// DecisionTree_38362647888633
// MI455X (gfx1250) — matrix operand order repaired and checked by reading, not compiled, not run on hardware
//
#include <hip/hip_runtime.h>
#include <hip/hip_bf16.h>

typedef __attribute__((ext_vector_type(16))) __bf16 v16bf;
typedef __attribute__((ext_vector_type(8)))  __bf16 v8bf;
typedef __attribute__((ext_vector_type(8)))  float   v8f;
typedef __attribute__((ext_vector_type(4)))  float   f4;

#define DIM      1024
#define NTOK     4096                 // B*S
#define NLEAF    16
#define NSTEPS   (NLEAF * DIM / 32)   // 512 real k-steps of 32 (+1 bias step)
#define TILE_M   128
#define TILE_N   128
#define LDSTR    40                   // padded LDS row stride (bf16 elements)

union ABfrag { v16bf v; v8bf h[2]; };
struct StageRegs { f4 a[4]; f4 b[4]; float sc; };

// ---------------------------------------------------------------------------
// Kernel 1: leaf probabilities. Only decision nodes 7..14 feed the output
// (each leaf is weighted by its direct parent only). One block per token,
// wave w (of 8) handles node 7+w: 1024-long dot product, wave32 butterfly
// reduction, sigmoid, write p and 1-p for the two children.
// ---------------------------------------------------------------------------
__global__ void __launch_bounds__(256) dtree_probs(
    const float* __restrict__ x,
    const float* __restrict__ dec_w,
    const float* __restrict__ dec_b,
    float* __restrict__ probs) {
  const int token = blockIdx.x;
  const int wave  = threadIdx.x >> 5;
  const int lane  = threadIdx.x & 31;
  const int node  = 7 + wave;

  const float* xr = x + (size_t)token * DIM;
  const float* wr = dec_w + (size_t)node * DIM;

  float s = 0.f;
#pragma unroll
  for (int j = 0; j < DIM / 32; ++j)
    s += xr[lane + 32 * j] * wr[lane + 32 * j];

#pragma unroll
  for (int off = 16; off > 0; off >>= 1)
    s += __shfl_xor(s, off, 32);

  if (lane == 0) {
    const float z = s + dec_b[node];
    const float p = 1.f / (1.f + __expf(-z));
    probs[token * NLEAF + 2 * wave]     = p;        // left child (even leaf)
    probs[token * NLEAF + 2 * wave + 1] = 1.f - p;  // right child (odd leaf)
  }
}

// ---------------------------------------------------------------------------
// Kernel 2: fused weighted mixture-of-leaves GEMM.
//   out[t,o] = sum_{l,d} (p[t,l]*x[t,d]) * W[l,o,d] + sum_l p[t,l]*b[l,o]
// One GEMM: M=4096, N=1024, K=16*1024 (+1 bias k-step), bf16 WMMA, f32 acc.
// Double-buffered LDS, pipeline per step: global loads (k+1) -> 8 WMMAs (k)
// -> cvt+ds_store (k+1) -> single barrier. Bias step peeled out of the loop.
// ---------------------------------------------------------------------------
__global__ void __launch_bounds__(256) dtree_gemm(
    const float* __restrict__ x,
    const float* __restrict__ leaf_w,
    const float* __restrict__ leaf_b,
    const float* __restrict__ probs,
    float* __restrict__ out) {
  __shared__ __bf16 At[2][TILE_M * LDSTR];
  __shared__ __bf16 Bt[2][TILE_N * LDSTR];

  const int tid  = threadIdx.x;
  const int lane = tid & 31;
  const int wave = tid >> 5;
  const int wm   = wave >> 2;                 // 0..1 -> 64-row band
  const int wn   = wave & 3;                  // 0..3 -> 32-col band
  const int wgM  = (int)blockIdx.x * TILE_M;
  const int wgN  = (int)blockIdx.y * TILE_N;

  // LDS fill assignment: thread t fills half of row t/2 (16 k-elements).
  const int fr = tid >> 1;                    // 0..127
  const int fo = (tid & 1) * 16;              // k-offset 0 or 16

  // Fragment addressing (documented 16-bit layouts):
  // A: lane m holds row m&15; VGPRs0-3 = K {0..7 | 8..15}, VGPRs4-7 = +16.
  const int aRow = lane & 15;
  const int aCh  = (lane >> 4) << 3;          // element offset 0 or 8
  // B: lane n holds column n&15; same K order as A: {0..7 | 8..15}, +16.
  const int bCh  = (lane >> 4) << 3;          // element offset 0 or 8

  v8f acc[4][2];
#pragma unroll
  for (int i = 0; i < 4; ++i)
#pragma unroll
    for (int j = 0; j < 2; ++j)
      acc[i][j] = {};

  // ---- pipeline stages -----------------------------------------------------
  auto stage_load = [&](int s, StageRegs& r) {
    const int l     = s >> 5;                       // leaf index
    const int dbase = ((s & 31) << 5) + fo;
    const int token = wgM + fr;
    r.sc = probs[token * NLEAF + l];
    const f4* sa = (const f4*)(x + (size_t)token * DIM + dbase);
    const f4* sb = (const f4*)(leaf_w + ((size_t)l * DIM + (wgN + fr)) * DIM + dbase);
#pragma unroll
    for (int c = 0; c < 4; ++c) { r.a[c] = sa[c]; r.b[c] = sb[c]; }
  };

  auto stage_load_bias = [&](StageRegs& r) {        // k-cols 0..15 = bias fold
    r.sc = 1.f;
#pragma unroll
    for (int c = 0; c < 4; ++c) { r.a[c] = {}; r.b[c] = {}; }
    if (fo == 0) {
      const f4* sp = (const f4*)(probs + (wgM + fr) * NLEAF);
#pragma unroll
      for (int c = 0; c < 4; ++c) r.a[c] = sp[c];
      const int o = wgN + fr;
#pragma unroll
      for (int c = 0; c < 4; ++c)
#pragma unroll
        for (int e = 0; e < 4; ++e)
          r.b[c][e] = leaf_b[(size_t)(4 * c + e) * DIM + o];
    }
  };

  auto stage_store = [&](const StageRegs& r, int buf) {
    __bf16* da = &At[buf][fr * LDSTR + fo];
    __bf16* db = &Bt[buf][fr * LDSTR + fo];
#pragma unroll
    for (int c = 0; c < 2; ++c) {
      v8bf oa, ob;
#pragma unroll
      for (int e = 0; e < 4; ++e) {
        oa[e]     = (__bf16)(r.a[2 * c][e]     * r.sc);
        oa[e + 4] = (__bf16)(r.a[2 * c + 1][e] * r.sc);
        ob[e]     = (__bf16)(r.b[2 * c][e]);
        ob[e + 4] = (__bf16)(r.b[2 * c + 1][e]);
      }
      *(v8bf*)(da + 8 * c) = oa;
      *(v8bf*)(db + 8 * c) = ob;
    }
  };

  auto compute = [&](int buf) {
    ABfrag af[4];
#pragma unroll
    for (int i = 0; i < 4; ++i) {
      const __bf16* rp = &At[buf][(wm * 64 + i * 16 + aRow) * LDSTR];
      af[i].h[0] = *(const v8bf*)(rp + aCh);
      af[i].h[1] = *(const v8bf*)(rp + 16 + aCh);
    }
    ABfrag bfm[2];
#pragma unroll
    for (int j = 0; j < 2; ++j) {
      const __bf16* rp = &Bt[buf][(wn * 32 + j * 16 + (lane & 15)) * LDSTR + bCh];
      bfm[j].h[0] = *(const v8bf*)(rp);
      bfm[j].h[1] = *(const v8bf*)(rp + 16);
    }
#pragma unroll
    for (int i = 0; i < 4; ++i)
#pragma unroll
      for (int j = 0; j < 2; ++j)
        acc[i][j] = __builtin_amdgcn_wmma_f32_16x16x32_bf16(
            false, af[i].v, false, bfm[j].v, (short)0, acc[i][j], false, false);
  };

  // ---- prologue: stage step 0 into buffer 0 --------------------------------
  StageRegs r;
  stage_load(0, r);
  stage_store(r, 0);
  __syncthreads();

  // ---- main pipeline: 512 steps, one barrier each --------------------------
  for (int s = 0; s < NSTEPS; ++s) {
    const int cur = s & 1;
    if (s + 1 < NSTEPS) stage_load(s + 1, r);   // global loads in flight...
    else                stage_load_bias(r);
    compute(cur);                               // ...hidden behind 8 WMMAs
    stage_store(r, cur ^ 1);                    // cvt + ds_store next buffer
    __syncthreads();
  }

  // ---- peeled bias step (staged into buffer 0 by the last iteration) -------
  compute(0);

  // ---- epilogue: write fp32 output -----------------------------------------
  // C layout: lanes 0-15 -> N=lane, M=vgpr; lanes 16-31 -> N=lane-16, M=vgpr+8
  const int rOff = (lane >> 4) * 8;
  const int col0 = wgN + wn * 32 + (lane & 15);
#pragma unroll
  for (int i = 0; i < 4; ++i) {
    const int row0 = wgM + wm * 64 + i * 16 + rOff;
#pragma unroll
    for (int j = 0; j < 2; ++j) {
      const int col = col0 + j * 16;
#pragma unroll
      for (int v = 0; v < 8; ++v)
        out[(size_t)(row0 + v) * DIM + col] = acc[i][j][v];
    }
  }
}

// ---------------------------------------------------------------------------
extern "C" void kernel_launch(void* const* d_in, const int* in_sizes, int n_in,
                              void* d_out, int out_size, void* d_ws, size_t ws_size,
                              hipStream_t stream) {
  const float* x      = (const float*)d_in[0];
  const float* dec_w  = (const float*)d_in[1];
  const float* dec_b  = (const float*)d_in[2];
  const float* leaf_w = (const float*)d_in[3];
  const float* leaf_b = (const float*)d_in[4];
  float* out   = (float*)d_out;
  float* probs = (float*)d_ws;   // 4096*16 fp32 = 256 KB scratch

  dtree_probs<<<NTOK, 256, 0, stream>>>(x, dec_w, dec_b, probs);

  dim3 grid(NTOK / TILE_M, DIM / TILE_N);   // 32 x 8 workgroups
  dtree_gemm<<<grid, 256, 0, stream>>>(x, leaf_w, leaf_b, probs, out);
}
